// NNConv_DX_DG_DR_77747497992604
// MI455X (gfx1250) — matrix operand order repaired and checked by reading, not compiled, not run on hardware
//
#include <hip/hip_runtime.h>

typedef __attribute__((ext_vector_type(16))) _Float16 v16h;
typedef __attribute__((ext_vector_type(8)))  float    v8f;
typedef __attribute__((address_space(3)))    _Float16 lds_h;

#define N_NODES 24576
#define E_EDGES 49152
#define B_GRAPH 768
#define ED 16
#define HB 4   // h-slices per LDS chunk (chunk = HB*4096 halfs = 32KB)

// ---------------------------------------------------------------------------
// gfx1250 async global->LDS copy (ASYNCcnt-tracked, bypasses VGPRs)
// ---------------------------------------------------------------------------
__device__ __forceinline__ void async_b128(unsigned lds_addr, unsigned goff,
                                           const void* sbase) {
    asm volatile("global_load_async_to_lds_b128 %0, %1, %2"
                 :
                 : "v"(lds_addr), "v"(goff), "s"(sbase)
                 : "memory");
}
__device__ __forceinline__ void wait_async0() {
    asm volatile("s_wait_asynccnt 0x0" ::: "memory");
}

// ---------------------------------------------------------------------------
// Pack Wb [64, 64*64] (+ bias bb[64*64] as 65th h-slice) into the WMMA
// B-fragment layout: Bpk[h][s][t][lane][j], 16 halfs per lane per frag.
// B matrix 32x16 (KxN), contiguous k order used here (matches A): lane l -> N = l%16, K = 16*(l/16) + j.
// k = h*64 + i, i = s*32 + K, n = t*16 + l%16.
// ---------------------------------------------------------------------------
__global__ void pack_B_kernel(const float* __restrict__ Wb,
                              const float* __restrict__ bb,
                              _Float16* __restrict__ Bpk) {
    int idx = blockIdx.x * blockDim.x + threadIdx.x;
    if (idx >= 65 * 4096) return;
    int h   = idx >> 12;
    int rem = idx & 4095;
    int j = rem & 15;
    int l = (rem >> 4) & 31;
    int t = (rem >> 9) & 3;
    int s = rem >> 11;
    int n = t * 16 + (l & 15);
    int K = 16 * (l >> 4) + j;
    int i = s * 32 + K;
    float v = (h < 64) ? Wb[h * 4096 + i * 64 + n] : bb[i * 64 + n];
    Bpk[idx] = (_Float16)v;
}

// h = relu(edge_attr @ Wa + ba)   [E,16] @ [16,64]
__global__ void edge_hidden_kernel(const float* __restrict__ ea,
                                   const float* __restrict__ Wa,
                                   const float* __restrict__ ba,
                                   float* __restrict__ h) {
    int idx = blockIdx.x * blockDim.x + threadIdx.x;
    if (idx >= E_EDGES * 64) return;
    int e = idx >> 6, o = idx & 63;
    const float* er = ea + (size_t)e * ED;
    float s = ba[o];
#pragma unroll
    for (int i = 0; i < ED; ++i) s += er[i] * Wa[i * 64 + o];
    h[idx] = fmaxf(s, 0.0f);
}

__global__ void count_deg_kernel(const int* __restrict__ dst, int* __restrict__ cnt) {
    int e = blockIdx.x * blockDim.x + threadIdx.x;
    if (e < E_EDGES) atomicAdd(&cnt[dst[e]], 1);
}

// ---------------------------------------------------------------------------
// Fused edge GEMM: per 16-edge tile, msg = A @ W' with
// A[e,(h,i)] = hedge[e,h] * xfeat[src[e],i], built on the fly as
// (scalar h per row) x (cached x-fragment).  Atomic scatter-add to dst.
// Block = 256 threads = 8 waves; each wave owns one 16-edge tile.
// W' chunks are double-buffered in LDS via async global->LDS copies so the
// staging overlaps the WMMA pipeline.
// ---------------------------------------------------------------------------
__global__ __launch_bounds__(256) void nnconv_gemm_kernel(
    const float* __restrict__ xfeat,       // [N,64]
    const int* __restrict__ eidx,          // [2,E]
    const float* __restrict__ hedge,       // [E,64]
    const _Float16* __restrict__ Bpk,      // [65][2][4][32][16]
    float* __restrict__ nodesum)           // [N,64] (atomic accum)
{
    __shared__ __align__(32) _Float16 sB[2 * HB * 4096]; // 2 x 32 KB W' chunks
    __shared__ __align__(32) _Float16 sH[8 * 64 * 16];   // 16 KB hedge tiles

    const int tid  = threadIdx.x;
    const int wave = tid >> 5;
    const int l    = tid & 31;
    const int g    = l >> 4;
    const int e0wg = blockIdx.x * 128;

    const unsigned sB_lds = (unsigned)(uintptr_t)(lds_h*)sB;
    const unsigned CHUNK_BYTES = HB * 4096 * 2;  // 32768

    // issue async load of chunk 0 into buffer 0 (8 x b128 per thread)
#pragma unroll
    for (int k = 0; k < 8; ++k) {
        unsigned c = (unsigned)tid + k * 256u;
        async_b128(sB_lds + c * 16u, c * 16u, Bpk);
    }

    // stage hedge tiles -> sH[wave][h][e%16]  (coalesced read, transposed store)
    for (int idx = tid; idx < 128 * 64; idx += 256) {
        int er = idx >> 6;
        int h  = idx & 63;
        sH[(er >> 4) * 1024 + h * 16 + (er & 15)] =
            (_Float16)hedge[(size_t)(e0wg + er) * 64 + h];
    }

    // build per-tile A-layout x fragments (one per i0 in {0,32})
    // A 16x32 frag, contiguous k order used here: lane l, elem j -> K = 16g + j
    const int em   = e0wg + wave * 16 + (l & 15);
    const int srow = eidx[em];
    const float* xr = xfeat + (size_t)srow * 64;
    v16h xa[2];
#pragma unroll
    for (int s = 0; s < 2; ++s)
#pragma unroll
        for (int j = 0; j < 16; ++j) {
            int i = s * 32 + 16 * g + j;
            xa[s][j] = (_Float16)xr[i];
        }

    v8f acc[4] = {};
    wait_async0();
    __syncthreads();

    const _Float16* hrow = &sH[wave * 1024];
    for (int hb = 0; hb < 64 / HB; ++hb) {
        const int cur = hb & 1;
        // prefetch next chunk into the other buffer (overlaps the WMMAs below)
        if (hb + 1 < 64 / HB) {
            unsigned gbase = (unsigned)(hb + 1) * CHUNK_BYTES;
            unsigned lbase = sB_lds + (unsigned)(cur ^ 1) * CHUNK_BYTES;
#pragma unroll
            for (int k = 0; k < 8; ++k) {
                unsigned c = (unsigned)tid + k * 256u;
                async_b128(lbase + c * 16u, gbase + c * 16u, Bpk);
            }
        }
        const _Float16* sBc = sB + (size_t)cur * (HB * 4096);
#pragma unroll
        for (int hh = 0; hh < HB; ++hh) {
            _Float16 hs = hrow[(hb * HB + hh) * 16 + (l & 15)];
#pragma unroll
            for (int s = 0; s < 2; ++s) {
                v16h am;
#pragma unroll
                for (int j = 0; j < 16; ++j) am[j] = xa[s][j] * hs;
                const v16h* bfr = (const v16h*)sBc;
#pragma unroll
                for (int t = 0; t < 4; ++t) {
                    v16h b = bfr[((hh * 2 + s) * 4 + t) * 32 + l];
                    acc[t] = __builtin_amdgcn_wmma_f32_16x16x32_f16(
                        false, am, false, b, (short)0, acc[t], false, false);
                }
            }
        }
        wait_async0();      // next chunk resident before anyone reads it
        __syncthreads();    // and everyone is done with the current buffer
    }

    // bias slice (h==64): A = x fragment directly
    {
        const v16h* bias = (const v16h*)(Bpk + (size_t)64 * 4096);
#pragma unroll
        for (int s = 0; s < 2; ++s)
#pragma unroll
            for (int t = 0; t < 4; ++t) {
                v16h b = bias[(s * 4 + t) * 32 + l];
                acc[t] = __builtin_amdgcn_wmma_f32_16x16x32_f16(
                    false, xa[s], false, b, (short)0, acc[t], false, false);
            }
    }

    // scatter-add: C/D layout: VGPR r, lane l -> row M = r + 8*(l>>4), col = t*16 + l%16
    const int* dste = eidx + E_EDGES;
#pragma unroll
    for (int rr = 0; rr < 8; ++rr) {
        int er = e0wg + wave * 16 + rr + 8 * g;
        int dd = dste[er];
        float* np = nodesum + (size_t)dd * 64 + (l & 15);
#pragma unroll
        for (int t = 0; t < 4; ++t) atomicAdd(np + t * 16, acc[t][rr]);
    }
}

// out = relu(sum/max(cnt,1) + xin @ root + bias)
__global__ void node_update_kernel(const float* __restrict__ xin,
                                   const float* __restrict__ sums,
                                   const int* __restrict__ cnt,
                                   const float* __restrict__ root,
                                   const float* __restrict__ bias,
                                   float* __restrict__ out) {
    int idx = blockIdx.x * blockDim.x + threadIdx.x;
    if (idx >= N_NODES * 64) return;
    int n = idx >> 6, o = idx & 63;
    float c = (float)max(cnt[n], 1);
    float s = sums[idx] / c + bias[o];
    const float* xrow = xin + (size_t)n * 64;
#pragma unroll 8
    for (int i = 0; i < 64; ++i) s += xrow[i] * root[i * 64 + o];
    out[idx] = fmaxf(s, 0.0f);
}

__device__ __forceinline__ int enc_key(float f) {
    int b = __float_as_int(f);
    return b ^ ((b >> 31) & 0x7fffffff);
}
__device__ __forceinline__ float dec_key(int k) {
    int b = k ^ ((k >> 31) & 0x7fffffff);
    return __int_as_float(b);
}

__global__ void init_gmax_kernel(int* __restrict__ gmaxkey) {
    int b = blockIdx.x * blockDim.x + threadIdx.x;
    if (b < B_GRAPH) gmaxkey[b] = 0x807FFFFF;  // enc_key(-inf)
}

__global__ void gate_kernel(const float* __restrict__ hn, const float* __restrict__ Wg,
                            const float* __restrict__ bg, const int* __restrict__ batch,
                            float* __restrict__ gate, int* __restrict__ gmaxkey) {
    int n = blockIdx.x * blockDim.x + threadIdx.x;
    if (n >= N_NODES) return;
    const float* hr = hn + (size_t)n * 64;
    float s = bg[0];
#pragma unroll 8
    for (int i = 0; i < 64; ++i) s += hr[i] * Wg[i];
    gate[n] = s;
    atomicMax(&gmaxkey[batch[n]], enc_key(s));
}

__global__ void expdenom_kernel(float* __restrict__ gate, const int* __restrict__ gmaxkey,
                                const int* __restrict__ batch, float* __restrict__ denom) {
    int n = blockIdx.x * blockDim.x + threadIdx.x;
    if (n >= N_NODES) return;
    int b = batch[n];
    float e = expf(gate[n] - dec_key(gmaxkey[b]));
    gate[n] = e;
    atomicAdd(&denom[b], e);
}

__global__ void pooled_kernel(const float* __restrict__ gate, const float* __restrict__ denom,
                              const float* __restrict__ hn, const int* __restrict__ batch,
                              float* __restrict__ pooled) {
    int idx = blockIdx.x * blockDim.x + threadIdx.x;
    if (idx >= N_NODES * 64) return;
    int n = idx >> 6, o = idx & 63;
    int b = batch[n];
    float alpha = gate[n] / denom[b];
    atomicAdd(&pooled[(size_t)b * 64 + o], alpha * hn[idx]);
}

__global__ void final_kernel(const float* __restrict__ pooled, const float* __restrict__ gg,
                             const float* __restrict__ rr, const float* __restrict__ Wfc,
                             const float* __restrict__ bfc, float* __restrict__ out) {
    int b = blockIdx.x * blockDim.x + threadIdx.x;
    if (b >= B_GRAPH) return;
    float s = bfc[0];
#pragma unroll 8
    for (int i = 0; i < 64; ++i)  s += pooled[(size_t)b * 64 + i] * Wfc[i];
#pragma unroll 8
    for (int i = 0; i < 128; ++i) s += gg[(size_t)b * 128 + i] * Wfc[64 + i];
#pragma unroll 8
    for (int i = 0; i < 128; ++i) s += rr[(size_t)b * 128 + i] * Wfc[192 + i];
    out[b] = s;
}

extern "C" void kernel_launch(void* const* d_in, const int* in_sizes, int n_in,
                              void* d_out, int out_size, void* d_ws, size_t ws_size,
                              hipStream_t stream) {
    (void)in_sizes; (void)n_in; (void)out_size; (void)ws_size;
    const float* x     = (const float*)d_in[0];
    const int*   eidx  = (const int*)  d_in[1];
    const float* eattr = (const float*)d_in[2];
    const int*   batch = (const int*)  d_in[3];
    const float* g     = (const float*)d_in[4];
    const float* r     = (const float*)d_in[5];
    const float* W1a = (const float*)d_in[6],  *b1a = (const float*)d_in[7];
    const float* W1b = (const float*)d_in[8],  *b1b = (const float*)d_in[9];
    const float* root1 = (const float*)d_in[10], *bias1 = (const float*)d_in[11];
    const float* W2a = (const float*)d_in[12], *b2a = (const float*)d_in[13];
    const float* W2b = (const float*)d_in[14], *b2b = (const float*)d_in[15];
    const float* root2 = (const float*)d_in[16], *bias2 = (const float*)d_in[17];
    const float* Wg  = (const float*)d_in[18], *bg  = (const float*)d_in[19];
    const float* Wfc = (const float*)d_in[20], *bfc = (const float*)d_in[21];
    float* out = (float*)d_out;

    // workspace carve-up (256B aligned slices)
    char* ws = (char*)d_ws;
    size_t off = 0;
    auto carve = [&](size_t bytes) { char* p = ws + off; off = (off + bytes + 255) & ~(size_t)255; return p; };
    float*    h1e   = (float*)   carve((size_t)E_EDGES * 64 * 4);
    float*    h2e   = (float*)   carve((size_t)E_EDGES * 64 * 4);
    _Float16* Bpk1  = (_Float16*)carve((size_t)65 * 4096 * 2);
    _Float16* Bpk2  = (_Float16*)carve((size_t)65 * 4096 * 2);
    float*    hn1   = (float*)   carve((size_t)N_NODES * 64 * 4);
    float*    hn2   = (float*)   carve((size_t)N_NODES * 64 * 4);
    float*    nsum  = (float*)   carve((size_t)N_NODES * 64 * 4);
    int*      cnt   = (int*)     carve((size_t)N_NODES * 4);
    float*    gate  = (float*)   carve((size_t)N_NODES * 4);
    int*      gmk   = (int*)     carve((size_t)B_GRAPH * 4);
    float*    denom = (float*)   carve((size_t)B_GRAPH * 4);
    float*    pooled= (float*)   carve((size_t)B_GRAPH * 64 * 4);

    hipMemsetAsync(cnt,    0, (size_t)N_NODES * 4, stream);
    hipMemsetAsync(denom,  0, (size_t)B_GRAPH * 4, stream);
    hipMemsetAsync(pooled, 0, (size_t)B_GRAPH * 64 * 4, stream);
    hipMemsetAsync(nsum,   0, (size_t)N_NODES * 64 * 4, stream);

    // weight packing + edge MLPs + degrees
    pack_B_kernel<<<(65 * 4096 + 255) / 256, 256, 0, stream>>>(W1b, b1b, Bpk1);
    pack_B_kernel<<<(65 * 4096 + 255) / 256, 256, 0, stream>>>(W2b, b2b, Bpk2);
    edge_hidden_kernel<<<(E_EDGES * 64) / 256, 256, 0, stream>>>(eattr, W1a, b1a, h1e);
    edge_hidden_kernel<<<(E_EDGES * 64) / 256, 256, 0, stream>>>(eattr, W2a, b2a, h2e);
    count_deg_kernel<<<E_EDGES / 256, 256, 0, stream>>>(eidx + E_EDGES, cnt);

    // layer 1
    nnconv_gemm_kernel<<<E_EDGES / 128, 256, 0, stream>>>(x, eidx, h1e, Bpk1, nsum);
    node_update_kernel<<<(N_NODES * 64) / 256, 256, 0, stream>>>(x, nsum, cnt, root1, bias1, hn1);

    // layer 2
    hipMemsetAsync(nsum, 0, (size_t)N_NODES * 64 * 4, stream);
    nnconv_gemm_kernel<<<E_EDGES / 128, 256, 0, stream>>>(hn1, eidx, h2e, Bpk2, nsum);
    node_update_kernel<<<(N_NODES * 64) / 256, 256, 0, stream>>>(hn1, nsum, cnt, root2, bias2, hn2);

    // attentional pooling + final fc
    init_gmax_kernel<<<(B_GRAPH + 255) / 256, 256, 0, stream>>>(gmk);
    gate_kernel<<<N_NODES / 256, 256, 0, stream>>>(hn2, Wg, bg, batch, gate, gmk);
    expdenom_kernel<<<N_NODES / 256, 256, 0, stream>>>(gate, gmk, batch, denom);
    pooled_kernel<<<(N_NODES * 64) / 256, 256, 0, stream>>>(gate, denom, hn2, batch, pooled);
    final_kernel<<<(B_GRAPH + 255) / 256, 256, 0, stream>>>(pooled, g, r, Wfc, bfc, out);
}
